// LongShortTermMemoryModel_14053132992924
// MI455X (gfx1250) — matrix operand order repaired and checked by reading, not compiled, not run on hardware
//
#include <hip/hip_runtime.h>

// LSTM fused kernel for MI455X (gfx1250, wave32, WMMA).
// T=1024, B=512, I=13, H=128. Batch-parallel: 32 workgroups x 16 batch rows.
// Recurrent GEMM per step: [16 x 160] x [160 x 512] via v_wmma_f32_16x16x32_f16,
// with x_t and biases folded into the K dimension (K = 128 h + 13 x + 1 bias + pad).
// Nonlinearities use v_rcp_f32/v_exp_f32 (TRANS ops co-execute with WMMA).

#define T_STEPS 1024
#define BATCH   512
#define NIN     13
#define HID     128
#define NKF     5            // K fragments of 32 (K = 160)
#define BS      16           // batch rows per workgroup
#define HSTR    168          // hbuf row stride in halfs (336B, 16B-aligned rows)
#define HFSTR   132          // f32 h row stride

typedef __attribute__((ext_vector_type(16))) _Float16 v16h;
typedef __attribute__((ext_vector_type(8)))  _Float16 v8h;
typedef __attribute__((ext_vector_type(8)))  float    v8f;
typedef __attribute__((ext_vector_type(4)))  float    v4f;

// Fast nonlinearities: single v_exp_f32 + v_rcp_f32 (no IEEE div sequences).
static __device__ __forceinline__ float sigm(float x) {
    return __builtin_amdgcn_rcpf(1.0f + __expf(-x));
}
static __device__ __forceinline__ float tanh_fast(float x) {
    float e = __expf(-2.0f * x);
    return (1.0f - e) * __builtin_amdgcn_rcpf(1.0f + e);
}

__global__ __launch_bounds__(256, 1)
void lstm_fused(const float* __restrict__ x,    // [T,B,13]
                const float* __restrict__ W_ih, // [512,13]
                const float* __restrict__ W_hh, // [512,128]
                const float* __restrict__ b_ih, // [512]
                const float* __restrict__ b_hh, // [512]
                const float* __restrict__ W_fc, // [7,128]
                const float* __restrict__ b_fc, // [7]
                float* __restrict__ out)        // [T*B,7] softmax
{
    __shared__ __align__(16) _Float16 hbuf[BS][HSTR];   // augmented h' in f16
    __shared__ __align__(16) float    hf[BS][HFSTR];    // f32 h for FC
    __shared__ __align__(16) float    wfc[7][HID];
    __shared__ float bfc[7];
    __shared__ float logits[BS][8];

    const int tid  = threadIdx.x;      // 0..255 (8 waves)
    const int wave = tid >> 5;         // 0..7 -> hidden cols 16w..16w+15
    const int lane = tid & 31;
    const int lm   = lane & 15;        // row (A) / column (B/C) index in tile
    const int lg   = (lane >> 4) & 1;  // lane group
    const int b0   = blockIdx.x * BS;  // batch slice base

    // ---- init LDS ----
    for (int i = tid; i < BS * HSTR; i += 256)
        ((_Float16*)hbuf)[i] = (_Float16)0.0f;          // h0 = 0, pads = 0
    for (int i = tid; i < 7 * HID; i += 256)
        ((float*)wfc)[i] = W_fc[i];
    if (tid < 7) bfc[tid] = b_fc[tid];
    __syncthreads();
    if (tid < BS) hbuf[tid][HID + NIN] = (_Float16)1.0f; // bias activation col 141

    // ---- preload B-matrix fragments of W' = [W_hh | W_ih | b] (loop-invariant) ----
    // Wave w owns gate tiles {s*8 + w : s in 0..3}; lane lm = gate column within tile.
    // B layout (32x16 f16): lanes 0-15 hold K=0..15, lanes 16-31 hold K=16..31,
    // element e in fragment = K offset (16*lg + e) within the 32-wide k-slab.
    v16h Bf[4][NKF];
#pragma unroll
    for (int s = 0; s < 4; ++s) {
        const int gc = s * HID + wave * 16 + lm;         // row of W_ih / W_hh (0..511)
        const float bias = b_ih[gc] + b_hh[gc];
        const float* wh = W_hh + (size_t)gc * HID;
        const float* wi = W_ih + (size_t)gc * NIN;
#pragma unroll
        for (int kt = 0; kt < NKF; ++kt) {
#pragma unroll
            for (int e = 0; e < 16; ++e) {
                const int k = kt * 32 + lg * 16 + e;
                float v;
                if (k < HID)            v = wh[k];
                else if (k < HID + NIN) v = wi[k - HID];
                else if (k == HID + NIN) v = bias;
                else                    v = 0.0f;
                Bf[s][kt][e] = (_Float16)v;
            }
        }
    }

    // cell state: lane holds c for rows m = 8*lg + r, hidden col j = 16*wave + lm
    float c[8];
#pragma unroll
    for (int r = 0; r < 8; ++r) c[r] = 0.0f;

    const int xm = tid / NIN;          // batch row for x staging (valid tid < 208)
    const int xi = tid - xm * NIN;
    const int j  = wave * 16 + lm;     // hidden column owned by this lane

    for (int t = 0; t < T_STEPS; ++t) {
        // ---- phase 0: stage x_t into augmented columns 128..140 ----
        const float* xt = x + ((size_t)t * BATCH + b0) * NIN;
        if (tid < BS * NIN)
            hbuf[xm][HID + xi] = (_Float16)xt[xm * NIN + xi];
        __syncthreads();

        // ---- phase 1: load A fragments (h'_{t-1}) from LDS ----
        // A layout (16x32 f16): lane m=lm; lg=0 holds K {0..15}, lg=1 holds {16..31}
        v16h a[NKF];
#pragma unroll
        for (int kt = 0; kt < NKF; ++kt) {
            const v8h lo = *(const v8h*)&hbuf[lm][kt * 32 + 16 * lg];
            const v8h hi = *(const v8h*)&hbuf[lm][kt * 32 + 16 * lg + 8];
            a[kt] = __builtin_shufflevector(lo, hi,
                        0, 1, 2, 3, 4, 5, 6, 7, 8, 9, 10, 11, 12, 13, 14, 15);
        }
        __syncthreads();   // all hbuf reads done; safe to overwrite h later

        // ---- phase 2: gates = h' x W'^T  (4 tiles x 5 k-frags of WMMA) ----
        v8f acc[4];
#pragma unroll
        for (int s = 0; s < 4; ++s)
#pragma unroll
            for (int e = 0; e < 8; ++e) acc[s][e] = 0.0f;
#pragma unroll
        for (int kt = 0; kt < NKF; ++kt) {
#pragma unroll
            for (int s = 0; s < 4; ++s) {
                acc[s] = __builtin_amdgcn_wmma_f32_16x16x32_f16(
                    false, a[kt], false, Bf[s][kt],
                    (short)0, acc[s], false, false);
            }
        }

        // ---- phase 3: LSTM cell nonlinearity (registers only) ----
        float hvals[8];
#pragma unroll
        for (int r = 0; r < 8; ++r) {
            const float gi = sigm(acc[0][r]);
            const float gf = sigm(acc[1][r]);
            const float gg = tanh_fast(acc[2][r]);
            const float go = sigm(acc[3][r]);
            const float cn = gf * c[r] + gi * gg;
            c[r] = cn;
            hvals[r] = go * tanh_fast(cn);
        }
        // write h_t back: f16 for next-step WMMA, f32 for FC
#pragma unroll
        for (int r = 0; r < 8; ++r) {
            const int m = lg * 8 + r;
            hbuf[m][j] = (_Float16)hvals[r];
            hf[m][j]   = hvals[r];
        }
        __syncthreads();

        // ---- phase 4: FC logits (16 rows x 7 classes = 112 dot-128) ----
        if (tid < BS * 7) {
            const int r  = tid / 7;
            const int cc = tid - r * 7;
            v4f av = {0.0f, 0.0f, 0.0f, 0.0f};
#pragma unroll
            for (int k4 = 0; k4 < HID / 4; ++k4) {
                const v4f hv = *(const v4f*)&hf[r][k4 * 4];
                const v4f wv = *(const v4f*)&wfc[cc][k4 * 4];
                av += hv * wv;
            }
            logits[r][cc] = bfc[cc] + av[0] + av[1] + av[2] + av[3];
        }
        __syncthreads();

        // ---- phase 5: softmax over 7 classes, write output ----
        if (tid < BS) {
            float l[7];
            float mx = -1e30f;
#pragma unroll
            for (int cc = 0; cc < 7; ++cc) {
                l[cc] = logits[tid][cc];
                mx = fmaxf(mx, l[cc]);
            }
            float ssum = 0.0f;
#pragma unroll
            for (int cc = 0; cc < 7; ++cc) {
                l[cc] = __expf(l[cc] - mx);
                ssum += l[cc];
            }
            const float inv = __builtin_amdgcn_rcpf(ssum);
            float* op = out + ((size_t)t * BATCH + b0 + tid) * 7;
#pragma unroll
            for (int cc = 0; cc < 7; ++cc) op[cc] = l[cc] * inv;
        }
        // next iteration's x-staging touches only hbuf[.][128..140], ordered by
        // the phase-0 barrier; no conflict with FC/softmax readers (hf/logits).
    }
}

extern "C" void kernel_launch(void* const* d_in, const int* in_sizes, int n_in,
                              void* d_out, int out_size, void* d_ws, size_t ws_size,
                              hipStream_t stream) {
    (void)in_sizes; (void)n_in; (void)d_ws; (void)ws_size; (void)out_size;
    const float* x    = (const float*)d_in[0];
    const float* W_ih = (const float*)d_in[1];
    const float* W_hh = (const float*)d_in[2];
    const float* b_ih = (const float*)d_in[3];
    const float* b_hh = (const float*)d_in[4];
    const float* W_fc = (const float*)d_in[5];
    const float* b_fc = (const float*)d_in[6];
    float* out = (float*)d_out;

    lstm_fused<<<dim3(BATCH / BS), dim3(256), 0, stream>>>(
        x, W_ih, W_hh, b_ih, b_hh, W_fc, b_fc, out);
}
